// MPNN_21131239096634
// MI455X (gfx1250) — matrix operand order repaired and checked by reading, not compiled, not run on hardware
//
#include <hip/hip_runtime.h>

typedef __attribute__((ext_vector_type(16))) _Float16 v16h;
typedef __attribute__((ext_vector_type(8)))  float    v8f;

#define NEG_SLOPE 0.01f
#define ECK_WAVES 8
#define SMEM_W2_HALFS (32*2*32*16)   /* [c][t][lane][j] = 32768 halves = 64KB */
#define SMEM_B2_HALFS (2*32*16)      /* [t][lane][j]    = 1024 halves = 2KB  */
#define H1_STRIDE 34                 /* pad to kill LDS bank conflicts       */
#define H1_SLOT   (16*H1_STRIDE)     /* 544 halves per wave                  */
#define SMEM_BYTES (SMEM_W2_HALFS*2 + SMEM_B2_HALFS*2 + ECK_WAVES*H1_SLOT*2 + 16*32*4 + 32*4)

__device__ __forceinline__ float lrelu(float v) { return v > 0.0f ? v : NEG_SLOPE * v; }

// ---------------------------------------------------------------- elementwise
__global__ void zero_kernel(float* __restrict__ p, int n) {
    int i = blockIdx.x * blockDim.x + threadIdx.x;
    if (i < n) p[i] = 0.0f;
}

__global__ void lrelu_kernel(const float* __restrict__ in, float* __restrict__ out, int n) {
    int i = blockIdx.x * blockDim.x + threadIdx.x;
    if (i < n) out[i] = lrelu(in[i]);
}

// ------------------------------------------------- h = lrelu(x @ W[64,32] + b)
__global__ __launch_bounds__(256) void node_fc_kernel(
    const float* __restrict__ x, const float* __restrict__ w,
    const float* __restrict__ b, float* __restrict__ h, int N)
{
    __shared__ float sw[64 * 32];
    __shared__ float sb[32];
    for (int i = threadIdx.x; i < 64 * 32; i += 256) sw[i] = w[i];
    if (threadIdx.x < 32) sb[threadIdx.x] = b[threadIdx.x];
    __syncthreads();
    int node = blockIdx.x * 256 + threadIdx.x;
    if (node >= N) return;
    float xr[64];
    #pragma unroll
    for (int i = 0; i < 64; ++i) xr[i] = x[(size_t)node * 64 + i];
    #pragma unroll 4
    for (int o = 0; o < 32; ++o) {
        float a = sb[o];
        #pragma unroll
        for (int i = 0; i < 64; ++i) a += xr[i] * sw[i * 32 + o];
        h[(size_t)node * 32 + o] = lrelu(a);
    }
}

// --------------------------------------------- acc = h @ root[32,32] + bias
__global__ __launch_bounds__(256) void root_bias_kernel(
    const float* __restrict__ h, const float* __restrict__ root,
    const float* __restrict__ bias, float* __restrict__ acc, int N)
{
    __shared__ float sr[32 * 32];
    __shared__ float sb[32];
    for (int i = threadIdx.x; i < 32 * 32; i += 256) sr[i] = root[i];
    if (threadIdx.x < 32) sb[threadIdx.x] = bias[threadIdx.x];
    __syncthreads();
    int node = blockIdx.x * 256 + threadIdx.x;
    if (node >= N) return;
    float hr[32];
    #pragma unroll
    for (int i = 0; i < 32; ++i) hr[i] = h[(size_t)node * 32 + i];
    #pragma unroll 4
    for (int o = 0; o < 32; ++o) {
        float a = sb[o];
        #pragma unroll
        for (int i = 0; i < 32; ++i) a += hr[i] * sr[i * 32 + o];
        acc[(size_t)node * 32 + o] = a;
    }
}

// -----------------------------------------------------------------------------
// Fused NNConv edge path: per 16-edge tile (one wave32):
//   h1 = relu(edge_attr @ w1 + b1)                 (VALU, tiny)
//   msg = u @ W2flat + hs @ B2mat                  (66x v_wmma_f32_16x16x32_f16)
//     where u[m,(k,i)] = h1[m,k]*hs[m,i] -> rank-1 A fragments per K-chunk,
//     built with packed v_pk_mul_f16 (vector*scalar splat)
//   atomicAdd msg into agg[dst]                    (global_atomic_add_f32)
// W2 never materialized per-edge (saves ~1.6 GB of HBM traffic per layer).
// -----------------------------------------------------------------------------
__global__ __launch_bounds__(256) void edge_conv_kernel(
    const float* __restrict__ h, const int* __restrict__ srcI, const int* __restrict__ dstI,
    const float* __restrict__ eattr,
    const float* __restrict__ w1, const float* __restrict__ b1,
    const float* __restrict__ w2, const float* __restrict__ b2,
    float* __restrict__ agg, int E)
{
    extern __shared__ char smem_raw[];
    _Float16* sW2 = (_Float16*)smem_raw;                  // 64 KB, B-fragment layout
    _Float16* sB2 = sW2 + SMEM_W2_HALFS;                  // 2 KB
    _Float16* sH1 = sB2 + SMEM_B2_HALFS;                  // per-wave h1 slabs
    float*    sw1 = (float*)(sH1 + ECK_WAVES * H1_SLOT);  // 2 KB
    float*    sb1 = sw1 + 16 * 32;                        // 128 B

    const int tid = threadIdx.x;
    // ---- stage edge-MLP weights (f16, pre-permuted into WMMA B layout) ----
    for (int i = tid; i < 16 * 32; i += 256) sw1[i] = w1[i];
    if (tid < 32) sb1[tid] = b1[tid];
    for (int i = tid; i < SMEM_B2_HALFS; i += 256) {
        int t = i >> 9, L = (i >> 4) & 31, j = i & 15;
        int hgs = L >> 4, ns = L & 15;
        sB2[i] = (_Float16)b2[(j + 16 * hgs) * 32 + 16 * t + ns];
    }
    for (int i = tid; i < SMEM_W2_HALFS; i += 256) {
        int c = i >> 10, r = i & 1023;
        int t = r >> 9, L = (r >> 4) & 31, j = r & 15;
        int hgs = L >> 4, ns = L & 15;
        sW2[i] = (_Float16)w2[c * 1024 + (j + 16 * hgs) * 32 + 16 * t + ns];
    }
    __syncthreads();

    const int lane = tid & 31;
    const int wid  = tid >> 5;
    const int m    = lane & 15;   // edge row within tile (A/M index)
    const int hg   = lane >> 4;   // half-wave group
    const int ncol = lane & 15;   // C/D column
    _Float16* myH1 = sH1 + wid * H1_SLOT;
    const v16h* w2frag = (const v16h*)sW2;
    const v16h* b2frag = (const v16h*)sB2;
    const int numTiles = (E + 15) >> 4;

    for (int tile = blockIdx.x * ECK_WAVES + wid; tile < numTiles; tile += gridDim.x * ECK_WAVES) {
        const int e0 = tile << 4;
        const int em = e0 + m;
        const bool mvalid = (em < E);
        const int eClamp = mvalid ? em : (E - 1);

        // ---- edge MLP layer 1 (each lane: row m, 16 hidden outputs) ----
        float ea[16];
        const float* eap = eattr + (size_t)eClamp * 16;
        #pragma unroll
        for (int i = 0; i < 16; ++i) ea[i] = mvalid ? eap[i] : 0.0f;
        #pragma unroll
        for (int kk = 0; kk < 16; ++kk) {
            int k = hg * 16 + kk;
            float a = sb1[k];
            #pragma unroll
            for (int i = 0; i < 16; ++i) a += ea[i] * sw1[i * 32 + k];
            myH1[m * H1_STRIDE + k] = (_Float16)(a > 0.0f ? a : 0.0f);
        }

        // ---- gather h[src] row, build permuted A fragment in registers ----
        // 16-bit A layout (contiguous, matches B staging): lane-lo K in {0..15}, lane-hi {16..31}
        int sidx = mvalid ? srcI[eClamp] : 0;
        const float* hp = h + (size_t)sidx * 32;
        v16h hsreg;
        #pragma unroll
        for (int j = 0; j < 16; ++j) {
            int i = j + 16 * hg;
            hsreg[j] = (_Float16)(mvalid ? hp[i] : 0.0f);
        }

        asm volatile("" ::: "memory");  // keep per-wave LDS h1 stores before the chunk-loop loads

        v8f acc0 = {0.f, 0.f, 0.f, 0.f, 0.f, 0.f, 0.f, 0.f};
        v8f acc1 = {0.f, 0.f, 0.f, 0.f, 0.f, 0.f, 0.f, 0.f};

        #pragma unroll 4
        for (int c = 0; c < 32; ++c) {
            _Float16 hc = myH1[m * H1_STRIDE + c];  // LDS broadcast read
            v16h a = hsreg * hc;                    // rank-1 A fragment: v_pk_mul_f16 x8
            v16h bf0 = w2frag[(c * 2 + 0) * 32 + lane];
            v16h bf1 = w2frag[(c * 2 + 1) * 32 + lane];
            acc0 = __builtin_amdgcn_wmma_f32_16x16x32_f16(false, a, false, bf0, (short)0, acc0, false, false);
            acc1 = __builtin_amdgcn_wmma_f32_16x16x32_f16(false, a, false, bf1, (short)0, acc1, false, false);
        }
        // bias term: msg += hs @ b2mat
        {
            v16h bb0 = b2frag[0 * 32 + lane];
            v16h bb1 = b2frag[1 * 32 + lane];
            acc0 = __builtin_amdgcn_wmma_f32_16x16x32_f16(false, hsreg, false, bb0, (short)0, acc0, false, false);
            acc1 = __builtin_amdgcn_wmma_f32_16x16x32_f16(false, hsreg, false, bb1, (short)0, acc1, false, false);
        }

        // ---- scatter-add into agg[dst] (C/D layout: lane n=L&15, M = p + 8*hg) ----
        #pragma unroll
        for (int p = 0; p < 8; ++p) {
            int mo = p + 8 * hg;
            int e = e0 + mo;
            if (e < E) {
                float* base = agg + (size_t)dstI[e] * 32;
                atomicAdd(base + ncol,      acc0[p]);
                atomicAdd(base + 16 + ncol, acc1[p]);
            }
        }
    }
}

// ----------------------------------------------------- segment-sum pooling
__global__ void pool_kernel(const float* __restrict__ atom, const int* __restrict__ batch,
                            float* __restrict__ pooled, int N)
{
    int idx = blockIdx.x * blockDim.x + threadIdx.x;
    if (idx >= N * 32) return;
    int node = idx >> 5, o = idx & 31;
    atomicAdd(&pooled[(size_t)batch[node] * 32 + o], atom[idx]);
}

// -------------------------------- L2-normalize + final FC: out = pn @ fcw + fcb
__global__ void finalize_kernel(const float* __restrict__ pooled, const float* __restrict__ fcw,
                                const float* __restrict__ fcb, float* __restrict__ out, int G)
{
    int g = blockIdx.x * blockDim.x + threadIdx.x;
    if (g >= G) return;
    float p[32];
    float ss = 0.f;
    #pragma unroll
    for (int i = 0; i < 32; ++i) { p[i] = pooled[(size_t)g * 32 + i]; ss += p[i] * p[i]; }
    float inv = 1.0f / fmaxf(sqrtf(ss), 1e-12f);
    #pragma unroll
    for (int i = 0; i < 32; ++i) p[i] *= inv;
    #pragma unroll 4
    for (int o = 0; o < 32; ++o) {
        float a = fcb[o];
        #pragma unroll
        for (int i = 0; i < 32; ++i) a += p[i] * fcw[i * 32 + o];
        out[(size_t)g * 32 + o] = a;
    }
}

// -----------------------------------------------------------------------------
extern "C" void kernel_launch(void* const* d_in, const int* in_sizes, int n_in,
                              void* d_out, int out_size, void* d_ws, size_t ws_size,
                              hipStream_t stream)
{
    (void)n_in; (void)ws_size;
    const float* x     = (const float*)d_in[0];
    const int*   eidx  = (const int*)d_in[1];
    const float* eattr = (const float*)d_in[2];
    const int*   batch = (const int*)d_in[3];
    // d_in[4] = num_graphs (device scalar; G derived from out_size instead)
    const float* nfc_w = (const float*)d_in[5];
    const float* nfc_b = (const float*)d_in[6];
    const float* e1w1  = (const float*)d_in[7];
    const float* e1b1  = (const float*)d_in[8];
    const float* e1w2  = (const float*)d_in[9];
    const float* e1b2  = (const float*)d_in[10];
    const float* root1 = (const float*)d_in[11];
    const float* bias1 = (const float*)d_in[12];
    const float* e2w1  = (const float*)d_in[13];
    const float* e2b1  = (const float*)d_in[14];
    const float* e2w2  = (const float*)d_in[15];
    const float* e2b2  = (const float*)d_in[16];
    const float* root2 = (const float*)d_in[17];
    const float* bias2 = (const float*)d_in[18];
    const float* fc_w  = (const float*)d_in[19];
    const float* fc_b  = (const float*)d_in[20];

    const int N = in_sizes[0] / 64;
    const int E = in_sizes[1] / 2;
    const int G = (out_size - N * 32) / 32;

    float* hbuf   = (float*)d_ws;                 // [N,32]
    float* accb   = hbuf + (size_t)N * 32;        // [N,32]
    float* pooled = accb + (size_t)N * 32;        // [G,32]
    float* outp   = (float*)d_out;                // [G,32]
    float* atom   = outp + (size_t)G * 32;        // [N,32]

    const int*   srcI = eidx;
    const int*   dstI = eidx + E;

    const int numTiles = (E + 15) / 16;
    int egBlocks = (numTiles + ECK_WAVES - 1) / ECK_WAVES;
    if (egBlocks > 2048) egBlocks = 2048;

    // h = lrelu(x @ nfc_w + nfc_b)
    node_fc_kernel<<<(N + 255) / 256, 256, 0, stream>>>(x, nfc_w, nfc_b, hbuf, N);

    // layer 1: acc = h@root1+bias1; acc += edge messages; h = lrelu(acc)
    root_bias_kernel<<<(N + 255) / 256, 256, 0, stream>>>(hbuf, root1, bias1, accb, N);
    edge_conv_kernel<<<egBlocks, 256, SMEM_BYTES, stream>>>(hbuf, srcI, dstI, eattr,
                                                            e1w1, e1b1, e1w2, e1b2, accb, E);
    lrelu_kernel<<<(N * 32 + 255) / 256, 256, 0, stream>>>(accb, hbuf, N * 32);

    // layer 2: atom_embs = lrelu(acc2) directly into d_out tail
    root_bias_kernel<<<(N + 255) / 256, 256, 0, stream>>>(hbuf, root2, bias2, accb, N);
    edge_conv_kernel<<<egBlocks, 256, SMEM_BYTES, stream>>>(hbuf, srcI, dstI, eattr,
                                                            e2w1, e2b1, e2w2, e2b2, accb, E);
    lrelu_kernel<<<(N * 32 + 255) / 256, 256, 0, stream>>>(accb, atom, N * 32);

    // pooled segment-sum -> L2 normalize -> final FC
    zero_kernel<<<(G * 32 + 255) / 256, 256, 0, stream>>>(pooled, G * 32);
    pool_kernel<<<(N * 32 + 255) / 256, 256, 0, stream>>>(atom, batch, pooled, N);
    finalize_kernel<<<(G + 127) / 128, 128, 0, stream>>>(pooled, fc_w, fc_b, outp, G);
}
